// MultiHeadSelfAttention_28561532518660
// MI455X (gfx1250) — matrix operand order repaired and checked by reading, not compiled, not run on hardware
//
#include <hip/hip_runtime.h>
#include <stdint.h>

#define B_SZ   1024
#define NT     65
#define C_DIM  768
#define H_NUM  12
#define D_DIM  64
#define L_DIM  256
#define NPAD   80
#define KPAD   96
#define M_ROWS (B_SZ * NT)       // 66560
#define BH     (B_SZ * H_NUM)    // 12288
#define NN2    (NT * NT)         // 4225

typedef __bf16 bf16;
typedef __attribute__((ext_vector_type(16))) __bf16 v16bf;
typedef __attribute__((ext_vector_type(8)))  float  v8f;

union AFrag { v16bf v; bf16 e[16]; };
union Acc   { v8f  v; float e[8];  };

__device__ __forceinline__ v8f wmma_bf16(v16bf a, v16bf b, v8f c) {
  return __builtin_amdgcn_wmma_f32_16x16x32_bf16(false, a, false, b, (short)0, c,
                                                 false, false);
}

// ---- fragment gathers (contiguous k order, same for A and B) --------------------------------
// A (16x32): lane m=L%16, half g=L/16; k = g*16+e
__device__ __forceinline__ v16bf load_a_frag(const bf16* t, int ld) {
  int lane = threadIdx.x & 31;
  int m = lane & 15, g = lane >> 4;
  AFrag f;
#pragma unroll
  for (int e = 0; e < 16; ++e) {
    int k = g * 16 + e;
    f.e[e] = t[m * ld + k];
  }
  return f.v;
}
__device__ __forceinline__ v16bf load_a_frag_f32(const float* t, int ld) {
  int lane = threadIdx.x & 31;
  int m = lane & 15, g = lane >> 4;
  AFrag f;
#pragma unroll
  for (int e = 0; e < 16; ++e) {
    int k = g * 16 + e;
    f.e[e] = (bf16)t[m * ld + k];
  }
  return f.v;
}
// B (32x16): lane n=L%16, half g=L/16; k = g*16+e.  Tile stored one output
// column's K-run per row: t[n*ld + k]
__device__ __forceinline__ v16bf load_b_frag(const bf16* t, int ld) {
  int lane = threadIdx.x & 31;
  int n = lane & 15, g = lane >> 4;
  AFrag f;
#pragma unroll
  for (int e = 0; e < 16; ++e) f.e[e] = t[n * ld + g * 16 + e];
  return f.v;
}
__device__ __forceinline__ v16bf load_b_frag_f32(const float* t, int ld) {
  int lane = threadIdx.x & 31;
  int n = lane & 15, g = lane >> 4;
  AFrag f;
#pragma unroll
  for (int e = 0; e < 16; ++e) f.e[e] = (bf16)t[n * ld + g * 16 + e];
  return f.v;
}
// B from K-major tile: t[k*ld + n]
__device__ __forceinline__ v16bf load_b_frag_km(const bf16* t, int ld) {
  int lane = threadIdx.x & 31;
  int n = lane & 15, g = lane >> 4;
  AFrag f;
#pragma unroll
  for (int e = 0; e < 16; ++e) f.e[e] = t[(g * 16 + e) * ld + n];
  return f.v;
}

// ---- CDNA5 async global->LDS DMA (ASYNCcnt) -------------------------------
__device__ __forceinline__ void async_ld_b128(uint32_t lds_byte_off, const void* g) {
  asm volatile("global_load_async_to_lds_b128 %0, %1, off"
               :: "v"(lds_byte_off), "v"(g) : "memory");
}
__device__ __forceinline__ void wait_async0() {
  asm volatile("s_wait_asynccnt 0" ::: "memory");
}
__device__ __forceinline__ uint32_t lds_off(const void* p) {
  return (uint32_t)(uintptr_t)p;
}

// ---------------------------------------------------------------------------
// Side network stage 1: t1 = silu(LN(cls @ sg_w1^T))        (B, 512)
// ---------------------------------------------------------------------------
__global__ void sg1_kernel(const float* __restrict__ x, const float* __restrict__ w1,
                           const float* __restrict__ g1, const float* __restrict__ b1,
                           float* __restrict__ t1) {
  __shared__ float cls[C_DIM];
  __shared__ float y[2 * L_DIM];
  __shared__ float rs[256], rss[256];
  int b = blockIdx.x, tid = threadIdx.x;
  const float* xr = x + (size_t)b * NT * C_DIM;  // token 0 row
#pragma unroll
  for (int i = tid; i < C_DIM; i += 256) cls[i] = xr[i];
  __syncthreads();
#pragma unroll
  for (int j = tid; j < 2 * L_DIM; j += 256) {
    const float* wr = w1 + (size_t)j * C_DIM;
    float a = 0.f;
    for (int k = 0; k < C_DIM; ++k) a += cls[k] * wr[k];
    y[j] = a;
  }
  __syncthreads();
  float s = 0.f, ss = 0.f;
#pragma unroll
  for (int j = tid; j < 2 * L_DIM; j += 256) { float v = y[j]; s += v; ss += v * v; }
  rs[tid] = s; rss[tid] = ss; __syncthreads();
  for (int st = 128; st > 0; st >>= 1) {
    if (tid < st) { rs[tid] += rs[tid + st]; rss[tid] += rss[tid + st]; }
    __syncthreads();
  }
  float mu = rs[0] / (2 * L_DIM);
  float var = rss[0] / (2 * L_DIM) - mu * mu;
  float inv = rsqrtf(var + 1e-5f);
#pragma unroll
  for (int j = tid; j < 2 * L_DIM; j += 256) {
    float v = (y[j] - mu) * inv * g1[j] + b1[j];
    v = v / (1.f + __expf(-v));
    t1[(size_t)b * 2 * L_DIM + j] = v;
  }
}

// ---------------------------------------------------------------------------
// Side network stage 2: lat = LN(t1 @ sg_w2^T)               (B, 3072)
// ---------------------------------------------------------------------------
__global__ void sg2_kernel(const float* __restrict__ t1, const float* __restrict__ w2,
                           const float* __restrict__ g2, const float* __restrict__ b2,
                           float* __restrict__ lat) {
  __shared__ float tin[2 * L_DIM];
  __shared__ float y[H_NUM * L_DIM];   // 3072 floats
  __shared__ float rs[256], rss[256];
  int b = blockIdx.x, tid = threadIdx.x;
#pragma unroll
  for (int i = tid; i < 2 * L_DIM; i += 256) tin[i] = t1[(size_t)b * 2 * L_DIM + i];
  __syncthreads();
#pragma unroll
  for (int j = tid; j < H_NUM * L_DIM; j += 256) {
    const float* wr = w2 + (size_t)j * (2 * L_DIM);
    float a = 0.f;
    for (int k = 0; k < 2 * L_DIM; ++k) a += tin[k] * wr[k];
    y[j] = a;
  }
  __syncthreads();
  float s = 0.f, ss = 0.f;
#pragma unroll
  for (int j = tid; j < H_NUM * L_DIM; j += 256) { float v = y[j]; s += v; ss += v * v; }
  rs[tid] = s; rss[tid] = ss; __syncthreads();
  for (int st = 128; st > 0; st >>= 1) {
    if (tid < st) { rs[tid] += rs[tid + st]; rss[tid] += rss[tid + st]; }
    __syncthreads();
  }
  float mu = rs[0] / (H_NUM * L_DIM);
  float var = rss[0] / (H_NUM * L_DIM) - mu * mu;
  float inv = rsqrtf(var + 1e-5f);
#pragma unroll
  for (int j = tid; j < H_NUM * L_DIM; j += 256)
    lat[(size_t)b * H_NUM * L_DIM + j] = (y[j] - mu) * inv * g2[j] + b2[j];
}

// ---------------------------------------------------------------------------
// Bias GEMM: biasOut(BH, 4225) = lat(BH,256) @ sg_bias_w(4225,256)^T   (WMMA,
// async double-buffered f32 tiles in LDS, bf16 convert at fragment gather)
// ---------------------------------------------------------------------------
#define LDA 36
__global__ void bias_gemm_kernel(const float* __restrict__ lat,
                                 const float* __restrict__ wbias,
                                 float* __restrict__ biasOut) {
  __shared__ float xa[2][32][LDA];
  __shared__ float wbt[2][64][LDA];
  int tid = threadIdx.x, wave = tid >> 5;
  int r0 = blockIdx.x * 32, c0 = blockIdx.y * 64;
  int wr = (wave >> 2) * 16, wc = (wave & 3) * 16;
  const char* ag = (const char*)(lat + (size_t)r0 * L_DIM);
  // per-thread DMA chunk coordinates
  int arow = tid >> 3, ach = tid & 7;                    // A: 256 x 16B
  int brow0 = tid >> 3, brow1 = (tid + 256) >> 3;        // B: 512 x 16B
  int bcol0 = c0 + brow0, bcol1 = c0 + brow1;
  if (bcol0 > NN2 - 1) bcol0 = NN2 - 1;                  // branchless clamp
  if (bcol1 > NN2 - 1) bcol1 = NN2 - 1;
  const char* bg0 = (const char*)(wbias + (size_t)bcol0 * L_DIM);
  const char* bg1 = (const char*)(wbias + (size_t)bcol1 * L_DIM);

  Acc acc;
#pragma unroll
  for (int i = 0; i < 8; ++i) acc.e[i] = 0.f;

#define ISSUE_TILES(buf, k0)                                                     \
  do {                                                                           \
    async_ld_b128(lds_off(&xa[buf][arow][ach * 4]),                              \
                  ag + (size_t)arow * (L_DIM * 4) + (k0) * 4 + ach * 16);        \
    async_ld_b128(lds_off(&wbt[buf][brow0][ach * 4]),                            \
                  bg0 + (k0) * 4 + ach * 16);                                    \
    async_ld_b128(lds_off(&wbt[buf][brow1][ach * 4]),                            \
                  bg1 + (k0) * 4 + ach * 16);                                    \
  } while (0)

  ISSUE_TILES(0, 0);
  wait_async0();
  __syncthreads();
  int cur = 0;
  for (int k0 = 0; k0 < L_DIM; k0 += 32) {
    if (k0 + 32 < L_DIM) ISSUE_TILES(cur ^ 1, k0 + 32);
    acc.v = wmma_bf16(load_a_frag_f32(&xa[cur][wr][0], LDA),
                      load_b_frag_f32(&wbt[cur][wc][0], LDA), acc.v);
    wait_async0();
    __syncthreads();
    cur ^= 1;
  }
#undef ISSUE_TILES
  int lane = tid & 31, n = lane & 15, g = lane >> 4;
#pragma unroll
  for (int r = 0; r < 8; ++r) {
    int row = r0 + wr + r + 8 * g;
    int col = c0 + wc + n;
    if (col < NN2) biasOut[(size_t)row * NN2 + col] = acc.e[r];
  }
}

// ---------------------------------------------------------------------------
// QKV GEMM + fused per-head LN of q/k (+ 1/sqrt(D) scale on q); writes padded
// bf16 q/k/v (B,H,NPAD,D).  grid=(M_ROWS/32, 36); slice s: part=s/12, h=s%12
// ---------------------------------------------------------------------------
__global__ void qkv_ln_kernel(const float* __restrict__ x, const float* __restrict__ qkv_w,
                              const float* __restrict__ qn_g, const float* __restrict__ qn_b,
                              const float* __restrict__ kn_g, const float* __restrict__ kn_b,
                              bf16* __restrict__ qb, bf16* __restrict__ kb,
                              bf16* __restrict__ vb) {
  __shared__ float xa[2][32][LDA];
  __shared__ float wbt[2][64][LDA];
  __shared__ float accs[32][65];
  int tid = threadIdx.x, wave = tid >> 5;
  int r0 = blockIdx.x * 32;
  int s = blockIdx.y;
  int part = s / H_NUM, h = s % H_NUM;
  int col0 = s * D_DIM;
  int wr = (wave >> 2) * 16, wc = (wave & 3) * 16;
  const char* ag = (const char*)(x + (size_t)r0 * C_DIM);
  const char* bg = (const char*)(qkv_w + (size_t)col0 * C_DIM);
  int arow = tid >> 3, ach = tid & 7;
  int brow0 = tid >> 3, brow1 = (tid + 256) >> 3;

  Acc acc;
#pragma unroll
  for (int i = 0; i < 8; ++i) acc.e[i] = 0.f;

#define ISSUE_TILES(buf, k0)                                                     \
  do {                                                                           \
    async_ld_b128(lds_off(&xa[buf][arow][ach * 4]),                              \
                  ag + (size_t)arow * (C_DIM * 4) + (k0) * 4 + ach * 16);        \
    async_ld_b128(lds_off(&wbt[buf][brow0][ach * 4]),                            \
                  bg + (size_t)brow0 * (C_DIM * 4) + (k0) * 4 + ach * 16);       \
    async_ld_b128(lds_off(&wbt[buf][brow1][ach * 4]),                            \
                  bg + (size_t)brow1 * (C_DIM * 4) + (k0) * 4 + ach * 16);       \
  } while (0)

  ISSUE_TILES(0, 0);
  wait_async0();
  __syncthreads();
  int cur = 0;
  for (int k0 = 0; k0 < C_DIM; k0 += 32) {
    if (k0 + 32 < C_DIM) ISSUE_TILES(cur ^ 1, k0 + 32);
    acc.v = wmma_bf16(load_a_frag_f32(&xa[cur][wr][0], LDA),
                      load_b_frag_f32(&wbt[cur][wc][0], LDA), acc.v);
    wait_async0();
    __syncthreads();
    cur ^= 1;
  }
#undef ISSUE_TILES
  {
    int lane = tid & 31, n = lane & 15, g = lane >> 4;
#pragma unroll
    for (int r = 0; r < 8; ++r) accs[wr + r + 8 * g][wc + n] = acc.e[r];
  }
  __syncthreads();
  if (tid < 32) {
    int rr = tid;
    int R = r0 + rr;
    int bb = R / NT, nn = R % NT;
    size_t base = (((size_t)bb * H_NUM + h) * NPAD + nn) * (size_t)D_DIM;
    if (part == 2) {
      for (int d = 0; d < D_DIM; ++d) vb[base + d] = (bf16)accs[rr][d];
    } else {
      float sm = 0.f, sq = 0.f;
      for (int d = 0; d < D_DIM; ++d) { float v = accs[rr][d]; sm += v; sq += v * v; }
      float mu = sm / D_DIM;
      float var = sq / D_DIM - mu * mu;
      float inv = rsqrtf(var + 1e-5f);
      if (part == 0) {
        for (int d = 0; d < D_DIM; ++d) {
          float v = ((accs[rr][d] - mu) * inv * qn_g[d] + qn_b[d]) * 0.125f;
          qb[base + d] = (bf16)v;
        }
      } else {
        for (int d = 0; d < D_DIM; ++d) {
          float v = (accs[rr][d] - mu) * inv * kn_g[d] + kn_b[d];
          kb[base + d] = (bf16)v;
        }
      }
    }
  }
}

// ---------------------------------------------------------------------------
// Attention per (b,h): S = qk^T + bias (masked), softmax, out = P @ V
// ---------------------------------------------------------------------------
#define LDQ (D_DIM + 8)   // 72
#define LDP (KPAD + 8)    // 104
__global__ void attn_kernel(const bf16* __restrict__ qb, const bf16* __restrict__ kb,
                            const bf16* __restrict__ vb, const float* __restrict__ bias,
                            bf16* __restrict__ ctx) {
  __shared__ bf16  qs[NPAD][LDQ];
  __shared__ bf16  ks[NPAD][LDQ];
  __shared__ bf16  vs[KPAD][LDQ];
  __shared__ float sc[NPAD][NPAD + 1];
  __shared__ bf16  ps[NPAD][LDP];
  int tid = threadIdx.x, wave = tid >> 5;
  int bh = blockIdx.x;
  int b = bh / H_NUM, h = bh % H_NUM;

  const char* qg = (const char*)(qb + (size_t)bh * NPAD * D_DIM);
  const char* kg = (const char*)(kb + (size_t)bh * NPAD * D_DIM);
  const char* vg = (const char*)(vb + (size_t)bh * NPAD * D_DIM);
  // 80 rows * 128B per row, 16B chunks -> 640 per tensor; straight-line issue
#pragma unroll
  for (int it = 0; it < 3; ++it) {
    int c = tid + it * 256;
    if (c < NPAD * 8) {
      int row = c >> 3, o = (c & 7) * 16;
      uint32_t loff = (uint32_t)(row * LDQ * 2 + o);
      size_t   goff = (size_t)row * 128 + o;
      async_ld_b128(lds_off(&qs[0][0]) + loff, qg + goff);
      async_ld_b128(lds_off(&ks[0][0]) + loff, kg + goff);
      async_ld_b128(lds_off(&vs[0][0]) + loff, vg + goff);
    }
  }
  // zero v pad rows 80..95 (K padded to 96 for the PV GEMM)
#pragma unroll
  for (int i = tid; i < 16 * LDQ; i += 256) {
    int rr = NPAD + i / LDQ, cc = i % LDQ;
    vs[rr][cc] = (bf16)0.0f;
  }
  wait_async0();
  __syncthreads();

  // ---- S = q @ k^T  (5x5 tiles of 16x16 over NPAD=80) + bias + mask ----
  for (int t = wave; t < 25; t += 8) {
    int i = t / 5, j = t % 5;
    Acc a;
#pragma unroll
    for (int r = 0; r < 8; ++r) a.e[r] = 0.f;
#pragma unroll
    for (int k0 = 0; k0 < D_DIM; k0 += 32) {
      v16bf af = load_a_frag(&qs[i * 16][k0], LDQ);
      v16bf bf_ = load_b_frag(&ks[j * 16][k0], LDQ);
      a.v = wmma_bf16(af, bf_, a.v);
    }
    int lane = tid & 31, n = lane & 15, g = lane >> 4;
#pragma unroll
    for (int r = 0; r < 8; ++r) {
      int row = i * 16 + r + 8 * g, col = j * 16 + n;
      float v = a.e[r];
      if (row < NT && col < NT)
        v += bias[((size_t)bh * NT + row) * NT + col];
      if (col >= NT) v = -1e30f;
      sc[row][col] = v;
    }
  }
  __syncthreads();

  // ---- softmax rows (one thread per row) ----
  if (tid < NPAD) {
    int row = tid;
    float m = -1e30f;
    for (int c = 0; c < NT; ++c) m = fmaxf(m, sc[row][c]);
    float s = 0.f;
    for (int c = 0; c < NT; ++c) { float e = __expf(sc[row][c] - m); sc[row][c] = e; s += e; }
    float inv = 1.f / s;
    for (int c = 0; c < KPAD; ++c)
      ps[row][c] = (c < NT) ? (bf16)(sc[row][c] * inv) : (bf16)0.0f;
  }
  __syncthreads();

  // ---- out = P @ V  (5x4 tiles of 16x16, K = 96) ----
  for (int t = wave; t < 20; t += 8) {
    int i = t / 4, j = t % 4;
    Acc a;
#pragma unroll
    for (int r = 0; r < 8; ++r) a.e[r] = 0.f;
#pragma unroll
    for (int k0 = 0; k0 < KPAD; k0 += 32) {
      v16bf af = load_a_frag(&ps[i * 16][k0], LDP);
      v16bf bf_ = load_b_frag_km(&vs[k0][j * 16], LDQ);
      a.v = wmma_bf16(af, bf_, a.v);
    }
    int lane = tid & 31, n = lane & 15, g = lane >> 4;
#pragma unroll
    for (int r = 0; r < 8; ++r) {
      int row = i * 16 + r + 8 * g, col = j * 16 + n;
      if (row < NT)
        ctx[((size_t)b * NT + row) * C_DIM + h * D_DIM + col] = (bf16)a.e[r];
    }
  }
}

// ---------------------------------------------------------------------------
// Output projection: out(M_ROWS,768) = ctx(bf16) @ proj_w(768,768)^T   (WMMA,
// async double-buffered; A tile stays bf16, B tile f32 converted at gather)
// ---------------------------------------------------------------------------
#define LDAB 40   // bf16 leading dim for A tile
__global__ void proj_kernel(const bf16* __restrict__ ctx, const float* __restrict__ pw,
                            float* __restrict__ out) {
  __shared__ bf16  xa[2][32][LDAB];
  __shared__ float wbt[2][64][LDA];
  int tid = threadIdx.x, wave = tid >> 5;
  int r0 = blockIdx.x * 32, c0 = blockIdx.y * 64;
  int wr = (wave >> 2) * 16, wc = (wave & 3) * 16;
  const char* ag = (const char*)(ctx + (size_t)r0 * C_DIM);
  const char* bg = (const char*)(pw + (size_t)c0 * C_DIM);
  // A: 32 rows x 64B = 128 chunks (threads 0..127); B: 512 chunks (2/thread)
  int arow = tid >> 2, ach = tid & 3;
  int brow0 = tid >> 3, brow1 = (tid + 256) >> 3;
  int bch = tid & 7;

  Acc acc;
#pragma unroll
  for (int i = 0; i < 8; ++i) acc.e[i] = 0.f;

#define ISSUE_TILES(buf, k0)                                                     \
  do {                                                                           \
    if (tid < 128)                                                               \
      async_ld_b128(lds_off(&xa[buf][arow][ach * 8]),                            \
                    ag + (size_t)arow * (C_DIM * 2) + (k0) * 2 + ach * 16);      \
    async_ld_b128(lds_off(&wbt[buf][brow0][bch * 4]),                            \
                  bg + (size_t)brow0 * (C_DIM * 4) + (k0) * 4 + bch * 16);       \
    async_ld_b128(lds_off(&wbt[buf][brow1][bch * 4]),                            \
                  bg + (size_t)brow1 * (C_DIM * 4) + (k0) * 4 + bch * 16);       \
  } while (0)

  ISSUE_TILES(0, 0);
  wait_async0();
  __syncthreads();
  int cur = 0;
  for (int k0 = 0; k0 < C_DIM; k0 += 32) {
    if (k0 + 32 < C_DIM) ISSUE_TILES(cur ^ 1, k0 + 32);
    acc.v = wmma_bf16(load_a_frag(&xa[cur][wr][0], LDAB),
                      load_b_frag_f32(&wbt[cur][wc][0], LDA), acc.v);
    wait_async0();
    __syncthreads();
    cur ^= 1;
  }
#undef ISSUE_TILES
  int lane = tid & 31, n = lane & 15, g = lane >> 4;
#pragma unroll
  for (int r = 0; r < 8; ++r) {
    int row = r0 + wr + r + 8 * g;
    int col = c0 + wc + n;
    out[(size_t)row * C_DIM + col] = acc.e[r];
  }
}

// ---------------------------------------------------------------------------
extern "C" void kernel_launch(void* const* d_in, const int* in_sizes, int n_in,
                              void* d_out, int out_size, void* d_ws, size_t ws_size,
                              hipStream_t stream) {
  (void)in_sizes; (void)n_in; (void)out_size; (void)ws_size;
  const float* x        = (const float*)d_in[0];
  const float* qkv_w    = (const float*)d_in[1];
  const float* proj_w   = (const float*)d_in[2];
  const float* qn_g     = (const float*)d_in[3];
  const float* qn_b     = (const float*)d_in[4];
  const float* kn_g     = (const float*)d_in[5];
  const float* kn_b     = (const float*)d_in[6];
  const float* sg_w1    = (const float*)d_in[7];
  const float* sg_ln1_g = (const float*)d_in[8];
  const float* sg_ln1_b = (const float*)d_in[9];
  const float* sg_w2    = (const float*)d_in[10];
  const float* sg_ln2_g = (const float*)d_in[11];
  const float* sg_ln2_b = (const float*)d_in[12];
  const float* sg_bias  = (const float*)d_in[13];

  char* ws = (char*)d_ws;
  size_t off = 0;
  auto alloc = [&](size_t bytes) -> void* {
    void* p = ws + off;
    off = (off + bytes + 255) & ~(size_t)255;
    return p;
  };
  const size_t qkvBytes = (size_t)BH * NPAD * D_DIM * sizeof(bf16);
  float* t1    = (float*)alloc((size_t)B_SZ * 2 * L_DIM * sizeof(float));
  float* lat   = (float*)alloc((size_t)B_SZ * H_NUM * L_DIM * sizeof(float));
  bf16*  qbuf  = (bf16*)alloc(qkvBytes);
  bf16*  kbuf  = (bf16*)alloc(qkvBytes);
  bf16*  vbuf  = (bf16*)alloc(qkvBytes);
  float* biasb = (float*)alloc((size_t)BH * NN2 * sizeof(float));
  bf16*  ctx   = (bf16*)alloc((size_t)M_ROWS * C_DIM * sizeof(bf16));

  // zero q/k/v so the padded rows (n in [65,80)) are valid zeros
  hipMemsetAsync(qbuf, 0, qkvBytes, stream);
  hipMemsetAsync(kbuf, 0, qkvBytes, stream);
  hipMemsetAsync(vbuf, 0, qkvBytes, stream);

  sg1_kernel<<<B_SZ, 256, 0, stream>>>(x, sg_w1, sg_ln1_g, sg_ln1_b, t1);
  sg2_kernel<<<B_SZ, 256, 0, stream>>>(t1, sg_w2, sg_ln2_g, sg_ln2_b, lat);
  bias_gemm_kernel<<<dim3(BH / 32, (NN2 + 63) / 64), 256, 0, stream>>>(lat, sg_bias, biasb);
  qkv_ln_kernel<<<dim3(M_ROWS / 32, 36), 256, 0, stream>>>(x, qkv_w, qn_g, qn_b,
                                                           kn_g, kn_b, qbuf, kbuf, vbuf);
  attn_kernel<<<BH, 256, 0, stream>>>(qbuf, kbuf, vbuf, biasb, ctx);
  proj_kernel<<<dim3(M_ROWS / 32, C_DIM / 64), 256, 0, stream>>>(ctx, proj_w,
                                                                 (float*)d_out);
}
